// Head_1194000908543
// MI455X (gfx1250) — matrix operand order repaired and checked by reading, not compiled, not run on hardware
//
#include <hip/hip_runtime.h>

// ---------------------------------------------------------------------------
// Attention head forward for MI455X (gfx1250, wave32, WMMA, async-to-LDS).
//   B=8, T=2048, E=1024, H=64
//   q=x@Wq (scale folded), k=x@Wk, v=x@Wv  -> bf16 in ws
//   flash-style causal softmax(q k^T) v    -> f32 out
// ---------------------------------------------------------------------------

typedef __attribute__((ext_vector_type(16))) __bf16 v16bf;
typedef __attribute__((ext_vector_type(8)))  __bf16 v8bf;
typedef __attribute__((ext_vector_type(8)))  float  v8f;

#define BATCH 8
#define TSEQ  2048
#define EMB   1024
#define HD    64
#define NEG_INF (-3.0e38f)

static __device__ __forceinline__ v8f wmma_bf16(v16bf a, v16bf b, v8f c) {
  // (neg_a, A, neg_b, B, c_mod, C, reuse_a, reuse_b)
  return __builtin_amdgcn_wmma_f32_16x16x32_bf16(false, a, false, b, (short)0, c,
                                                 false, false);
}

static __device__ __forceinline__ v16bf cat8(v8bf lo, v8bf hi) {
  return __builtin_shufflevector(lo, hi, 0, 1, 2, 3, 4, 5, 6, 7,
                                         8, 9, 10, 11, 12, 13, 14, 15);
}

// ---------------------------------------------------------------------------
// Kernel 0: WT[n][e] = W{q|k|v}[e][n%64] as bf16, N-major; Wq scaled by 1/8.
// ---------------------------------------------------------------------------
__global__ __launch_bounds__(256) void prep_w(const float* __restrict__ Wq,
                                              const float* __restrict__ Wk,
                                              const float* __restrict__ Wv,
                                              __bf16* __restrict__ WT) {
  int idx = blockIdx.x * 256 + threadIdx.x;   // over 192*1024
  if (idx >= 192 * EMB) return;
  int n = idx >> 10;            // 0..191
  int e = idx & (EMB - 1);      // 0..1023
  int h = n & 63;
  float v;
  if (n < 64)       v = Wq[e * HD + h] * 0.125f;   // fold 1/sqrt(H)
  else if (n < 128) v = Wk[e * HD + h];
  else              v = Wv[e * HD + h];
  WT[idx] = (__bf16)v;
}

// ---------------------------------------------------------------------------
// Kernel 1: QKV projection. One block = one 16-row tile of x (B*T rows).
// 4 waves; wave w handles N-tiles {w, w+4, w+8} of the 192 output columns.
// ---------------------------------------------------------------------------
__global__ __launch_bounds__(128) void qkv_proj(const float* __restrict__ x,
                                                const __bf16* __restrict__ WT,
                                                __bf16* __restrict__ qs,
                                                __bf16* __restrict__ ks,
                                                __bf16* __restrict__ vT) {
  __shared__ __align__(16) __bf16 xa[16 * EMB];   // 32 KB bf16 A tile

  const int row0 = blockIdx.x * 16;               // global token row
  const int tid  = threadIdx.x;

  // cooperative load + fp32->bf16 convert: 16*1024 elems = 4096 float4
  const float4* xs = (const float4*)(x + (size_t)row0 * EMB);
  for (int i = tid; i < 4096; i += 128) {
    float4 f = xs[i];
    int b4 = i * 4;
    xa[b4 + 0] = (__bf16)f.x;
    xa[b4 + 1] = (__bf16)f.y;
    xa[b4 + 2] = (__bf16)f.z;
    xa[b4 + 3] = (__bf16)f.w;
  }
  __syncthreads();

  const int wave = tid >> 5;
  const int lane = tid & 31;
  const int half = lane >> 4;
  const int n16  = lane & 15;
  const __bf16* arow = xa + (size_t)(lane & 15) * EMB;

  for (int nt = wave; nt < 12; nt += 4) {
    v8f acc = {};
    const __bf16* wrow = WT + (size_t)(nt * 16 + n16) * EMB;
#pragma unroll 4
    for (int k0 = 0; k0 < EMB; k0 += 32) {
      v8bf alo = *(const v8bf*)(arow + k0 + 8 * half);
      v8bf ahi = *(const v8bf*)(arow + k0 + 16 + 8 * half);
      v16bf a  = cat8(alo, ahi);
      v16bf b  = cat8(*(const v8bf*)(wrow + k0 + 8 * half),
                      *(const v8bf*)(wrow + k0 + 16 + 8 * half));
      acc = wmma_bf16(a, b, acc);
    }
    const int h = ((nt & 3) << 4) + n16;          // column within head dim
    if (nt < 4) {
#pragma unroll
      for (int r = 0; r < 8; ++r)
        qs[(size_t)(row0 + r + 8 * half) * HD + h] = (__bf16)acc[r];
    } else if (nt < 8) {
#pragma unroll
      for (int r = 0; r < 8; ++r)
        ks[(size_t)(row0 + r + 8 * half) * HD + h] = (__bf16)acc[r];
    } else {
#pragma unroll
      for (int r = 0; r < 8; ++r) {
        int tok = row0 + r + 8 * half;
        int bb  = tok >> 11;                      // /2048
        int tt  = tok & (TSEQ - 1);
        vT[((size_t)bb * HD + h) * TSEQ + tt] = (__bf16)acc[r];
      }
    }
  }
}

// ---------------------------------------------------------------------------
// Async copy helpers (gfx1250 GLOBAL_LOAD_ASYNC_TO_LDS, tracked by ASYNCcnt).
// Each instruction moves 32 lanes x 16B = 512B. 8 instructions = one 4KB tile.
// ---------------------------------------------------------------------------
static __device__ __forceinline__ void async_copy_k(unsigned ldsOff,
                                                    unsigned long long gbase,
                                                    int kB, int lane) {
  // K tile: 32 contiguous rows of 64 bf16 -> contiguous 4KB at kB*128 bytes.
#pragma unroll
  for (int j = 0; j < 8; ++j) {
    unsigned goff = (unsigned)kB * 128u + (unsigned)j * 512u + (unsigned)lane * 16u;
    unsigned loff = ldsOff + (unsigned)j * 512u + (unsigned)lane * 16u;
    asm volatile("global_load_async_to_lds_b128 %0, %1, %2"
                 :: "v"(loff), "v"(goff), "s"(gbase) : "memory");
  }
}

static __device__ __forceinline__ void async_copy_v(unsigned ldsOff,
                                                    unsigned long long gbase,
                                                    int kB, int lane) {
  // V tile: 64 h-rows x 64B (stride 4096B in vT) -> LDS [64][32] bf16.
  const int hrow = lane >> 2;          // 0..7
  const int chnk = lane & 3;           // 0..3 (16B chunks of a 64B row)
#pragma unroll
  for (int j = 0; j < 8; ++j) {
    int h = j * 8 + hrow;
    unsigned goff = (unsigned)h * 4096u + (unsigned)kB * 2u + (unsigned)chnk * 16u;
    unsigned loff = ldsOff + (unsigned)h * 64u + (unsigned)chnk * 16u;
    asm volatile("global_load_async_to_lds_b128 %0, %1, %2"
                 :: "v"(loff), "v"(goff), "s"(gbase) : "memory");
  }
}

// ---------------------------------------------------------------------------
// Kernel 2: flash attention, causal. One wave (block of 32) per 32-query
// tile (two 16-row A tiles sharing each K/V fragment -> 2x arithmetic
// intensity). KV tiles of 32 keys are double-buffered in LDS via async copy.
// Per KV tile: S = Q K^T (8 wmma), online softmax, O += P V (8 wmma).
// ---------------------------------------------------------------------------
__global__ __launch_bounds__(32) void attn(const __bf16* __restrict__ qs,
                                           const __bf16* __restrict__ ks,
                                           const __bf16* __restrict__ vT,
                                           float* __restrict__ out) {
  __shared__ __align__(16) __bf16 kbuf[2][32 * 64];   // 2 x 4KB K tiles
  __shared__ __align__(16) __bf16 vbuf[2][64 * 32];   // 2 x 4KB V tiles
  __shared__ __align__(16) __bf16 plds[32 * 40];      // P staging, 80B rows

  const int b     = blockIdx.x >> 6;                  // 64 q-tiles(32) per batch
  const int qBase = (blockIdx.x & 63) * 32;
  const int lane  = threadIdx.x;
  const int half  = lane >> 4;
  const int n16   = lane & 15;

  const __bf16* qb = qs + (size_t)b * TSEQ * HD;
  const unsigned long long kg = (unsigned long long)(const void*)(ks + (size_t)b * TSEQ * HD);
  const unsigned long long vg = (unsigned long long)(const void*)(vT + (size_t)b * HD * TSEQ);
  const unsigned kLds[2] = {(unsigned)(uintptr_t)&kbuf[0][0],
                            (unsigned)(uintptr_t)&kbuf[1][0]};
  const unsigned vLds[2] = {(unsigned)(uintptr_t)&vbuf[0][0],
                            (unsigned)(uintptr_t)&vbuf[1][0]};

  // ---- Q fragments (A layout), cached for the whole kernel ----
  v16bf aq[2][2];
#pragma unroll
  for (int rt = 0; rt < 2; ++rt) {
    const __bf16* qrow = qb + (size_t)(qBase + rt * 16 + (lane & 15)) * HD;
    aq[rt][0] = cat8(*(const v8bf*)(qrow + 8 * half),
                     *(const v8bf*)(qrow + 16 + 8 * half));
    aq[rt][1] = cat8(*(const v8bf*)(qrow + 32 + 8 * half),
                     *(const v8bf*)(qrow + 48 + 8 * half));
  }

  v8f o[2][4] = {};
  float mrun[2][8], lrun[2][8];
#pragma unroll
  for (int rt = 0; rt < 2; ++rt)
#pragma unroll
    for (int r = 0; r < 8; ++r) { mrun[rt][r] = NEG_INF; lrun[rt][r] = 0.0f; }

  const int kEnd = qBase + 32;
  // prologue: fetch first KV tile into buffer 0
  async_copy_k(kLds[0], kg, 0, lane);
  async_copy_v(vLds[0], vg, 0, lane);

  for (int kB = 0; kB < kEnd; kB += 32) {
    const int buf = (kB >> 5) & 1;
    if (kB + 32 < kEnd) {
      // prefetch next tile into the other buffer, then wait for current tile
      async_copy_k(kLds[buf ^ 1], kg, kB + 32, lane);
      async_copy_v(vLds[buf ^ 1], vg, kB + 32, lane);
      asm volatile("s_wait_asynccnt 16" ::: "memory");  // <=16 outstanding = next tile only
    } else {
      asm volatile("s_wait_asynccnt 0" ::: "memory");
    }

    const __bf16* kt = &kbuf[buf][0];
    const __bf16* vt = &vbuf[buf][0];

    // ---- S = Q K^T: two row tiles x two 16-key column tiles ----
    v16bf bk[2][2];
#pragma unroll
    for (int kn = 0; kn < 2; ++kn) {
      const __bf16* krow = kt + (size_t)(kn * 16 + n16) * 64;
      bk[kn][0] = cat8(*(const v8bf*)(krow + 8 * half),
                       *(const v8bf*)(krow + 16 + 8 * half));
      bk[kn][1] = cat8(*(const v8bf*)(krow + 32 + 8 * half),
                       *(const v8bf*)(krow + 48 + 8 * half));
    }
    v8f s[2][2];
#pragma unroll
    for (int rt = 0; rt < 2; ++rt)
#pragma unroll
      for (int kn = 0; kn < 2; ++kn) {
        v8f acc = {};
        acc = wmma_bf16(aq[rt][0], bk[kn][0], acc);
        acc = wmma_bf16(aq[rt][1], bk[kn][1], acc);
        s[rt][kn] = acc;
      }

    // ---- causal mask (only the diagonal tile kB == qBase needs it) ----
    if (kB == qBase) {
#pragma unroll
      for (int rt = 0; rt < 2; ++rt)
#pragma unroll
        for (int r = 0; r < 8; ++r) {
          int rloc = rt * 16 + r + 8 * half;      // row within the 32-query tile
          if (n16 > rloc)      s[rt][0][r] = NEG_INF;
          if (16 + n16 > rloc) s[rt][1][r] = NEG_INF;
        }
    }

    // ---- online softmax (row r+8*half of each tile lives in one 16-lane half) ----
#pragma unroll
    for (int rt = 0; rt < 2; ++rt)
#pragma unroll
      for (int r = 0; r < 8; ++r) {
        float v = fmaxf(s[rt][0][r], s[rt][1][r]);
        v = fmaxf(v, __shfl_xor(v, 1, 32));
        v = fmaxf(v, __shfl_xor(v, 2, 32));
        v = fmaxf(v, __shfl_xor(v, 4, 32));
        v = fmaxf(v, __shfl_xor(v, 8, 32));
        float mnew  = fmaxf(mrun[rt][r], v);
        float alpha = __expf(mrun[rt][r] - mnew);
        mrun[rt][r] = mnew;
        float p0 = __expf(s[rt][0][r] - mnew);
        float p1 = __expf(s[rt][1][r] - mnew);
        float rs = p0 + p1;
        rs += __shfl_xor(rs, 1, 32);
        rs += __shfl_xor(rs, 2, 32);
        rs += __shfl_xor(rs, 4, 32);
        rs += __shfl_xor(rs, 8, 32);
        lrun[rt][r] = lrun[rt][r] * alpha + rs;
        o[rt][0][r] *= alpha; o[rt][1][r] *= alpha;
        o[rt][2][r] *= alpha; o[rt][3][r] *= alpha;
        plds[(rt * 16 + r + 8 * half) * 40 + n16]      = (__bf16)p0;
        plds[(rt * 16 + r + 8 * half) * 40 + 16 + n16] = (__bf16)p1;
      }
    __syncthreads();   // IR fence; LDS is in-order within the single wave

    // ---- reload P in A layout (rows rt*16 + (lane&15)) ----
    v16bf pa[2];
#pragma unroll
    for (int rt = 0; rt < 2; ++rt) {
      const __bf16* prow = plds + (size_t)(rt * 16 + (lane & 15)) * 40;
      pa[rt] = cat8(*(const v8bf*)(prow + 8 * half),
                    *(const v8bf*)(prow + 16 + 8 * half));
    }
    __syncthreads();   // keep next iteration's stores after these loads

    // ---- O += P V over 4 head-dim tiles; V fragments shared by both row tiles ----
    v16bf bv[4];
#pragma unroll
    for (int t = 0; t < 4; ++t)
      bv[t] = cat8(*(const v8bf*)(vt + (size_t)(t * 16 + n16) * 32 + 8 * half),
                   *(const v8bf*)(vt + (size_t)(t * 16 + n16) * 32 + 16 + 8 * half));
#pragma unroll
    for (int rt = 0; rt < 2; ++rt)
#pragma unroll
      for (int t = 0; t < 4; ++t)
        o[rt][t] = wmma_bf16(pa[rt], bv[t], o[rt][t]);
  }

  // ---- epilogue: normalize and write f32 output [B][T][H] ----
#pragma unroll
  for (int rt = 0; rt < 2; ++rt)
#pragma unroll
    for (int r = 0; r < 8; ++r) {
      float inv = 1.0f / lrun[rt][r];
      float* p = out + ((size_t)b * TSEQ + qBase + rt * 16 + r + 8 * half) * HD + n16;
      p[0]  = o[rt][0][r] * inv;
      p[16] = o[rt][1][r] * inv;
      p[32] = o[rt][2][r] * inv;
      p[48] = o[rt][3][r] * inv;
    }
}

// ---------------------------------------------------------------------------
// Launch
// ---------------------------------------------------------------------------
extern "C" void kernel_launch(void* const* d_in, const int* in_sizes, int n_in,
                              void* d_out, int out_size, void* d_ws, size_t ws_size,
                              hipStream_t stream) {
  const float* x  = (const float*)d_in[0];
  const float* Wq = (const float*)d_in[1];
  const float* Wk = (const float*)d_in[2];
  const float* Wv = (const float*)d_in[3];
  float* out = (float*)d_out;

  char* ws = (char*)d_ws;
  __bf16* WT = (__bf16*)(ws);                       // 192*1024*2   =  384 KB
  __bf16* qs = (__bf16*)(ws + 393216);              // 16384*64*2   = 2048 KB
  __bf16* ks = (__bf16*)(ws + 2490368);             // 16384*64*2   = 2048 KB
  __bf16* vT = (__bf16*)(ws + 4587520);             // 8*64*2048*2  = 2048 KB

  prep_w<<<dim3((192 * 1024 + 255) / 256), dim3(256), 0, stream>>>(Wq, Wk, Wv, WT);
  qkv_proj<<<dim3(1024), dim3(128), 0, stream>>>(x, WT, qs, ks, vT);
  attn<<<dim3(512), dim3(32), 0, stream>>>(qs, ks, vT, out);
}
